// ObliviousDecisionTree_90142773608699
// MI455X (gfx1250) — matrix operand order repaired and checked by reading, not compiled, not run on hardware
//
#include <hip/hip_runtime.h>
#include <hip/hip_bf16.h>

// ---------------------------------------------------------------------------
// NODE oblivious-decision-tree forward, fused for gfx1250 (MI455X).
//   B=8192, F=256 (features), N=128 (trees), D=6 (depth), U=16, C=2^D=64
// Pipeline:
//   prep:  x -> f16, response -> f16, sparsemax(selectors) -> f16 (K-major)
//   main:  GEMM1 (8192x768x256, v_wmma_f32_16x16x32_f16)
//          -> gates -> factored leaf products -> GEMM2 (K=64 per tree, WMMA)
//          -> mean over trees
// ---------------------------------------------------------------------------

typedef __attribute__((ext_vector_type(8)))  _Float16 v8h;
typedef __attribute__((ext_vector_type(16))) _Float16 v16h;
typedef __attribute__((ext_vector_type(8)))  float    v8f;

#define BATCH  8192
#define FEAT   256
#define NTREE  128
#define DEPTH  6
#define NU     16
#define NLEAF  64
#define NDCOL  (NTREE * DEPTH)   // 768 columns of GEMM1

// ---------------------------------------------------------------------------
// f32 -> f16 elementwise convert
// ---------------------------------------------------------------------------
__global__ void cvt_f32_to_f16(const float* __restrict__ src,
                               _Float16* __restrict__ dst, int n) {
    int i = blockIdx.x * blockDim.x + threadIdx.x;
    if (i < n) dst[i] = (_Float16)src[i];
}

// ---------------------------------------------------------------------------
// sparsemax over depth (rows of 6) of feature_selection_logits (F,N,D),
// storing result transposed + f16 as selH[nd][i] (K-major for WMMA B frags).
// ---------------------------------------------------------------------------
__global__ void sparsemax6_kernel(const float* __restrict__ logits,
                                  _Float16* __restrict__ selH) {
    int r = blockIdx.x * blockDim.x + threadIdx.x;   // r = i*N + n
    if (r >= FEAT * NTREE) return;
    const int i = r >> 7;        // feature
    const int n = r & 127;       // tree

    float z[DEPTH], zs[DEPTH];
#pragma unroll
    for (int d = 0; d < DEPTH; ++d) { z[d] = logits[r * DEPTH + d]; zs[d] = z[d]; }

    // descending sort (fixed compare-exchange network via bubble passes)
#pragma unroll
    for (int a = 0; a < DEPTH; ++a)
#pragma unroll
        for (int b = 0; b < DEPTH - 1; ++b) {
            float lo = fminf(zs[b], zs[b + 1]);
            float hi = fmaxf(zs[b], zs[b + 1]);
            zs[b] = hi; zs[b + 1] = lo;
        }

    float cum = 0.f, csk = 0.f; int kmax = 1;
#pragma unroll
    for (int j = 0; j < DEPTH; ++j) {
        cum += zs[j];
        if (1.0f + (float)(j + 1) * zs[j] > cum) { kmax = j + 1; csk = cum; }
    }
    const float tau = (csk - 1.0f) / (float)kmax;

#pragma unroll
    for (int d = 0; d < DEPTH; ++d) {
        float v = z[d] - tau;
        v = v > 0.f ? v : 0.f;
        selH[(n * DEPTH + d) * FEAT + i] = (_Float16)v;   // [nd][k]
    }
}

// ---------------------------------------------------------------------------
// Fused main kernel. One block = 16 batch rows. 8 waves; wave w owns trees
// [w*16, w*16+16) == nd columns [w*96, w*96+96) == 6 WMMA column tiles.
//
// Fragment layout assumptions (CDNA5 ISA 7.12.2, wave32):
//  A (16-bit, 16x32): lane m=L%16; 8 VGPRs hold K = kb+(L/16)*16 .. +15,
//                     a contiguous k order chosen here, not the ISA table order.
//  B (16-bit, 32x16): lane n=L%16; 8 VGPRs hold K = kb+(L/16)*16 .. +15 (same order as A).
//  C (f32, 16x16):    VGPR v: M = v + 8*(L/16), N = L%16.
// ---------------------------------------------------------------------------
__global__ void __launch_bounds__(256)
node_main_kernel(const _Float16* __restrict__ xH,      // [B][F]   f16
                 const _Float16* __restrict__ selH,    // [ND][F]  f16
                 const _Float16* __restrict__ respH,   // [N][U][C] f16
                 const float*    __restrict__ thr,     // [N*D]
                 const float*    __restrict__ ltemp,   // [N*D]
                 float*          __restrict__ out) {   // [B][U]
    __shared__ float ldsG[8][16][96];   // gates: [wave][b_local][local nd]
    __shared__ float ldsR[8][256];      // per-wave partial output tiles

    const int w  = threadIdx.x >> 5;    // wave 0..7
    const int L  = threadIdx.x & 31;    // lane
    const int lh = L & 15;              // lane % 16
    const int s  = L >> 4;              // half-wave select
    const int b0 = blockIdx.x * 16;
    const int cbase = w * 96;           // wave's first nd column

    // ---------------- GEMM1: fv = x @ selectors (6 tiles, K=256) -----------
    v8f acc[6] = {};
    const _Float16* xrow = xH + (size_t)(b0 + lh) * FEAT;
#pragma unroll
    for (int kk = 0; kk < 8; ++kk) {
        const int kb = kk * 32;
        v8h a0 = *(const v8h*)(xrow + kb + s * 16);
        v8h a1 = *(const v8h*)(xrow + kb + s * 16 + 8);
        v16h afrag = __builtin_shufflevector(a0, a1,
            0,1,2,3,4,5,6,7,8,9,10,11,12,13,14,15);
#pragma unroll
        for (int t = 0; t < 6; ++t) {
            const int nd = cbase + t * 16 + lh;
            v16h bfrag = *(const v16h*)(selH + (size_t)nd * FEAT + kb + s * 16);
            acc[t] = __builtin_amdgcn_wmma_f32_16x16x32_f16(
                false, afrag, false, bfrag, (short)0, acc[t], false, false);
        }
    }

    // ------------- epilogue: thresholds, temperatures, gates -> LDS --------
#pragma unroll
    for (int t = 0; t < 6; ++t) {
        const int nd = cbase + t * 16 + lh;
        const float th = thr[nd];
        const float et = __expf(-ltemp[nd]);
#pragma unroll
        for (int v = 0; v < 8; ++v) {
            float tl = (acc[t][v] - th) * et;
            float g  = fminf(fmaxf(0.5f * tl + 0.5f, 0.0f), 1.0f);
            ldsG[w][v + 8 * s][t * 16 + lh] = g;   // rows per C layout
        }
    }
    __syncthreads();

    // ------------- GEMM2: leaf products @ response, 16 trees/wave ----------
    // Leaf value: rg(c) = prod_d ( bit_d(c) ? (1-g[d]) : g[d] ),
    // factored as pl[c&7] * ph[c>>3].
    v8f accO = {};
    for (int tn = 0; tn < 16; ++tn) {
        const int n = w * 16 + tn;
        float g[DEPTH];
#pragma unroll
        for (int d = 0; d < DEPTH; ++d) g[d] = ldsG[w][lh][tn * 6 + d];

        float pl[8], ph[8];
        {
            const float s0 = g[0], c0 = 1.f - g[0];
            const float s1 = g[1], c1 = 1.f - g[1];
            const float s2 = g[2], c2 = 1.f - g[2];
            float l2[4] = { s0 * s1, c0 * s1, s0 * c1, c0 * c1 };
#pragma unroll
            for (int j = 0; j < 4; ++j) { pl[j] = l2[j] * s2; pl[4 + j] = l2[j] * c2; }
            const float s3 = g[3], c3 = 1.f - g[3];
            const float s4 = g[4], c4 = 1.f - g[4];
            const float s5 = g[5], c5 = 1.f - g[5];
            float h2[4] = { s3 * s4, c3 * s4, s3 * c4, c3 * c4 };
#pragma unroll
            for (int j = 0; j < 4; ++j) { ph[j] = h2[j] * s5; ph[4 + j] = h2[j] * c5; }
        }

#pragma unroll
        for (int ks = 0; ks < 2; ++ks) {               // K = 64 = 2 x 32
            // lane's two 8-leaf runs: c = kb+s*16+j and c = kb+s*16+8+j
            const float ph0 = ph[ks * 4 + 2 * s];
            const float ph1 = ph[ks * 4 + 2 * s + 1];
            v16h afrag;
#pragma unroll
            for (int j = 0; j < 8; ++j) {
                afrag[j]     = (_Float16)(pl[j] * ph0);
                afrag[8 + j] = (_Float16)(pl[j] * ph1);
            }
            v16h bfrag = *(const v16h*)(respH + ((size_t)n * NU + lh) * NLEAF
                                        + ks * 32 + s * 16);
            accO = __builtin_amdgcn_wmma_f32_16x16x32_f16(
                false, afrag, false, bfrag, (short)0, accO, false, false);
        }
    }

    // ------------- cross-wave mean over trees, store ----------------------
#pragma unroll
    for (int v = 0; v < 8; ++v)
        ldsR[w][(v + 8 * s) * 16 + lh] = accO[v];
    __syncthreads();

    const int tid = threadIdx.x;        // tid -> (m = tid/16, u = tid%16)
    float sum = 0.f;
#pragma unroll
    for (int ww = 0; ww < 8; ++ww) sum += ldsR[ww][tid];
    out[(size_t)(b0 + (tid >> 4)) * NU + (tid & 15)] = sum * (1.0f / (float)NTREE);
}

// ---------------------------------------------------------------------------
extern "C" void kernel_launch(void* const* d_in, const int* in_sizes, int n_in,
                              void* d_out, int out_size, void* d_ws, size_t ws_size,
                              hipStream_t stream) {
    const float* x     = (const float*)d_in[0];   // (8192,256)
    const float* fsl   = (const float*)d_in[1];   // (256,128,6)
    const float* thr   = (const float*)d_in[2];   // (128,6)
    const float* ltemp = (const float*)d_in[3];   // (128,6)
    const float* resp  = (const float*)d_in[4];   // (128,16,64)
    // d_in[5] = depth (==6), compile-time constant here
    float* out = (float*)d_out;

    char* ws = (char*)d_ws;
    _Float16* xH    = (_Float16*)ws;                                   // 4 MiB
    _Float16* selH  = (_Float16*)(ws + (size_t)BATCH * FEAT * 2);      // 384 KiB
    _Float16* respH = (_Float16*)(ws + (size_t)BATCH * FEAT * 2
                                     + (size_t)NDCOL * FEAT * 2);      // 256 KiB

    const int nx = BATCH * FEAT;
    cvt_f32_to_f16<<<(nx + 255) / 256, 256, 0, stream>>>(x, xH, nx);
    const int nr = NTREE * NU * NLEAF;
    cvt_f32_to_f16<<<(nr + 255) / 256, 256, 0, stream>>>(resp, respH, nr);
    sparsemax6_kernel<<<(FEAT * NTREE + 255) / 256, 256, 0, stream>>>(fsl, selH);

    node_main_kernel<<<BATCH / 16, 256, 0, stream>>>(xH, selH, respH,
                                                     thr, ltemp, out);
}
